// Actor_81827716924053
// MI455X (gfx1250) — matrix operand order repaired and checked by reading, not compiled, not run on hardware
//
#include <hip/hip_runtime.h>
#include <stdint.h>

// ---------------------------------------------------------------------------
// MI455X (gfx1250, wave32) implementation of the Actor network.
// Dense layers -> v_wmma_f32_16x16x32_f16 tile GEMM (f16 in, f32 accum).
// GAT layers fuse dual-view masked softmax + aggregation in LDS; the 32KB
// value tile is fetched with the Tensor Data Mover (tensor_load_to_lds +
// s_wait_tensorcnt) when the builtin is available.
// Workspace requirement: ~278 MB.
// ---------------------------------------------------------------------------

typedef __attribute__((ext_vector_type(16))) _Float16 v16h;
typedef __attribute__((ext_vector_type(8)))  float    v8f;
typedef __attribute__((ext_vector_type(4)))  unsigned int u32x4;
typedef __attribute__((ext_vector_type(4)))  int      i32x4;
typedef __attribute__((ext_vector_type(8)))  int      i32x8;

#define BNROWS 65536      // B*N
#define NN     256
#define NNP    264        // padded alpha-tile row (bank rotate)

#if __has_builtin(__builtin_amdgcn_tensor_load_to_lds)
#define HAVE_TDM 1
#else
#define HAVE_TDM 0
#endif

// ---------------------------- workspace layout -----------------------------
#define OFF_OBS_B   ((size_t)0)                                  // [BN x 32] f16
#define OFF_XM      (OFF_OBS_B + (size_t)BNROWS*32*2)            // [BN x 96] f16
#define OFF_OBSF    (OFF_XM    + (size_t)BNROWS*96*2)            // [BN x 160] f16
#define OFF_MT2     (OFF_OBSF  + (size_t)BNROWS*160*2)           // [BN x 32] f16
#define OFF_BT2     (OFF_MT2   + (size_t)BNROWS*32*2)            // [BN x 32] f16
#define OFF_PAD_END (OFF_BT2   + (size_t)BNROWS*32*2)
#define OFF_WCOMP   (OFF_PAD_END)                                // [32 x 128]
#define OFF_WCOOP   (OFF_WCOMP + (size_t)32*128*2)               // [96 x 128]
#define OFF_W1H     (OFF_WCOOP + (size_t)96*128*2)               // [160 x 128]
#define OFF_W2H     (OFF_W1H   + (size_t)160*128*2)              // [128 x 128]
#define OFF_WIH     (OFF_W2H   + (size_t)128*128*2)              // [160 x 384]
#define OFF_WHRZ    (OFF_WIH   + (size_t)160*384*2)              // [128 x 256]
#define OFF_WHN     (OFF_WHRZ  + (size_t)128*256*2)              // [128 x 128]
#define OFF_MW1     (OFF_WHN   + (size_t)128*128*2)              // [128 x 32]
#define OFF_MW2     (OFF_MW1   + (size_t)128*32*2)               // [32 x 16]
#define OFF_MW3     (OFF_MW2   + (size_t)32*16*2)                // [32 x 128]
#define OFF_BW1     (OFF_MW3   + (size_t)32*128*2)               // [128 x 32]
#define OFF_BW2     (OFF_BW1   + (size_t)128*32*2)               // [32 x 16]
#define OFF_BIASC   (OFF_BW2   + (size_t)32*16*2)                // [384] f32
#define OFF_W_END   (OFF_BIASC + (size_t)384*4)
#define OFF_GI      ((OFF_W_END + 255) & ~(size_t)255)           // [BN x 384] f32
#define OFF_GHN     (OFF_GI    + (size_t)BNROWS*384*4)           // [BN x 128] f32
#define OFF_ES1     (OFF_GHN   + (size_t)BNROWS*128*4)
#define OFF_ED1     (OFF_ES1   + (size_t)BNROWS*4)
#define OFF_ES2     (OFF_ED1   + (size_t)BNROWS*4)
#define OFF_ED2     (OFF_ES2   + (size_t)BNROWS*4)
#define OFF_K1H     (OFF_ED2   + (size_t)BNROWS*4)               // [BN x 64] f16
#define OFF_K2H     (OFF_K1H   + (size_t)BNROWS*64*2)
#define OFF_HPREH   (OFF_K2H   + (size_t)BNROWS*64*2)            // [BN x 128] f16
#define OFF_H1H     (OFF_HPREH + (size_t)BNROWS*128*2)
#define OFF_COMH    (OFF_H1H   + (size_t)BNROWS*128*2)
#define OFF_HTH     (OFF_COMH  + (size_t)BNROWS*128*2)
#define OFF_MT1     (OFF_HTH   + (size_t)BNROWS*128*2)           // [BN x 32] f16
#define OFF_BT1     (OFF_MT1   + (size_t)BNROWS*32*2)
#define OFF_WH      (OFF_BT1   + (size_t)BNROWS*32*2)            // [BN x 128] f16

// --------------------- LDS fragment helpers (b128 path) --------------------
// A fragment (16-bit A, 16x32): halves 0..7 at [lm*ld + hi*8], 8..15 at +16.
__device__ __forceinline__ v16h load_a_frag(const _Float16* base, int ld,
                                            int lm, int hi, int k0)
{
    v16h a;
    *(uint4*)&a       = *(const uint4*)(base + lm * ld + k0 + (hi << 3));
    *((uint4*)&a + 1) = *(const uint4*)(base + lm * ld + k0 + 16 + (hi << 3));
    return a;
}
// B fragment (16-bit B, 32x16): lane holds column N = lm; halves 0..7 at K = hi*8+s, 8..15 at K = 16+hi*8+(s-8).
__device__ __forceinline__ v16h load_b_frag(const _Float16* base, int ld, int lm, int hi)
{
    v16h b;
#pragma unroll
    for (int s = 0; s < 16; ++s) {
        const int k = (s < 8) ? ((hi << 3) + s) : (16 + (hi << 3) + (s - 8));
        b[s] = base[k * ld + lm];
    }
    return b;
}

#if HAVE_TDM
// 2-D TDM load: rows x cols f16 tile, contiguous (stride == cols), into LDS.
__device__ __forceinline__ void tdm_load_tile_f16(void* lds, const void* gsrc,
                                                  unsigned rows, unsigned cols)
{
    const unsigned lds_off = (unsigned)(uintptr_t)lds;        // generic LDS addr[31:0]
    const uint64_t ga = (uint64_t)(uintptr_t)gsrc;
    u32x4 g0 = { 1u,                                          // count=1 (user D#)
                 lds_off,
                 (unsigned)ga,
                 (unsigned)((ga >> 32) & 0x1FFFFFFu) | (2u << 30) }; // type=2
    i32x8 g1 = { 0x00010000,                 // data_size=2B; no mask/pad/iterate
                 (int)(cols << 16),          // tensor_dim0[15:0] in bits 63:48
                 (int)(rows << 16),          // tensor_dim1[15:0] in bits 95:80
                 (int)(cols << 16),          // tile_dim0 in bits 127:112
                 (int)rows,                  // tile_dim1 in bits 143:128
                 (int)cols,                  // tensor_dim0_stride in bits 191:160
                 0, 0 };
    i32x4 gz = { 0, 0, 0, 0 };
#if defined(__clang_major__) && (__clang_major__ >= 23)
    i32x8 gz8 = { 0, 0, 0, 0, 0, 0, 0, 0 };
    __builtin_amdgcn_tensor_load_to_lds(g0, g1, gz, gz, gz8, 0);
#else
    __builtin_amdgcn_tensor_load_to_lds(g0, g1, gz, gz, 0);
#endif
#if __has_builtin(__builtin_amdgcn_s_wait_tensorcnt)
    __builtin_amdgcn_s_wait_tensorcnt(0);
#else
    asm volatile("s_wait_tensorcnt 0x0" ::: "memory");
#endif
}
#endif

// ---------------------------------------------------------------------------
// Generic WMMA GEMM: one wave per 16x16 output tile. A:[rows x K] f16 row-major,
// W:[K x N] f16 row-major. f32 accumulate (optional read-accumulate).
// act: 0=none 1=relu 2=tanh. Writes f32 and/or f16 outputs.
// ---------------------------------------------------------------------------
__global__ __launch_bounds__(32)
void gemm16(const _Float16* __restrict__ A, int lda,
            const _Float16* __restrict__ W, int ldw,
            const float* __restrict__ bias, int K,
            float* outF, int ldof, int accumulate,
            _Float16* outH, int ldoh, int colOffH, int act)
{
    __shared__ __align__(16) _Float16 sA[16 * 32];
    __shared__ __align__(16) _Float16 sB[32 * 16];
    const int lane = threadIdx.x;
    const int row0 = blockIdx.x * 16;
    const int col0 = blockIdx.y * 16;
    const int hi = lane >> 4, lm = lane & 15;

    v8f c = {};
    if (accumulate) {
#pragma unroll
        for (int v = 0; v < 8; ++v)
            c[v] = outF[(size_t)(row0 + v + (hi << 3)) * ldof + col0 + lm];
    }

    for (int k0 = 0; k0 < K; k0 += 32) {
        // A tile 16x32 halves = 64 x b128; B tile 32x16 halves = 64 x b128.
#pragma unroll
        for (int u = 0; u < 2; ++u) {
            const int ca = lane + 32 * u;
            const int ra = ca >> 2, cola = (ca & 3) << 3;
            ((uint4*)sA)[ca] =
                *(const uint4*)(A + (size_t)(row0 + ra) * lda + k0 + cola);
            const int cb = lane + 32 * u;
            const int rb = cb >> 1, colb = (cb & 1) << 3;
            ((uint4*)sB)[cb] =
                *(const uint4*)(W + (size_t)(k0 + rb) * ldw + col0 + colb);
        }
        __syncthreads();
        const v16h a = load_a_frag(sA, 32, lm, hi, 0);
        const v16h b = load_b_frag(sB, 16, lm, hi);
        c = __builtin_amdgcn_wmma_f32_16x16x32_f16(false, a, false, b,
                                                   (short)0, c, false, false);
        __syncthreads();
    }

    const float bv = bias ? bias[col0 + lm] : 0.0f;
#pragma unroll
    for (int v = 0; v < 8; ++v) {
        float x = c[v] + bv;
        if (act == 1)      x = fmaxf(x, 0.0f);
        else if (act == 2) x = tanhf(x);
        const size_t r = (size_t)(row0 + v + (hi << 3));
        const int cc = col0 + lm;
        if (outF) outF[r * ldof + cc] = x;
        if (outH) outH[r * ldoh + colOffH + cc] = (_Float16)x;
    }
}

// ---------------------------------------------------------------------------
// Fused dual-view GAT: per (batch, 16 dest cols): joint softmax over 2N=512
// incoming edges, then WMMA aggregation out[j,:] = sum_i (a_sta+a_req)[i,j]*k[i,:]
// (exploits v = concat([k,k])). 4 waves; LDS ~58 KB.
// ---------------------------------------------------------------------------
__global__ __launch_bounds__(128)
void gat_agg(const float* __restrict__ e_src, const float* __restrict__ e_dst,
             const float* __restrict__ distance, const float* __restrict__ a_edge,
             const int* __restrict__ adj_sta, const int* __restrict__ adj_req,
             const _Float16* __restrict__ kH,
             _Float16* out1, int ldo1, int off1,
             _Float16* out2, int ldo2, int off2)
{
    __shared__ __align__(16) _Float16 sK[NN * 64];     // values (32 KB)
    __shared__ _Float16 sS[NN * 16];                   // station-view scores
    __shared__ _Float16 sR[NN * 16];                   // request-view scores
    __shared__ __align__(16) _Float16 sWT[16 * NNP];   // alpha, transposed [j][i]
    __shared__ float    sEs[NN];
    __shared__ float    sEd[16];
    __shared__ float    red[16 * 8];

    const int b = blockIdx.y, j0 = blockIdx.x * 16;
    const int t = threadIdx.x;
    const size_t rowb = (size_t)b * NN;

    for (int i = t; i < NN; i += 128) sEs[i] = e_src[rowb + i];
    if (t < 16) sEd[t] = e_dst[rowb + j0 + t];

#if HAVE_TDM
    if (t < 32)                                       // wave 0 issues the DMA
        tdm_load_tile_f16(sK, kH + rowb * 64, NN, 64);
#else
    {   // fallback: 2048 x b128, unrolled
        const uint4* kg = (const uint4*)(kH + rowb * 64);
        uint4* ks = (uint4*)sK;
#pragma unroll
        for (int u = 0; u < 16; ++u) ks[t + 128 * u] = kg[t + 128 * u];
    }
#endif
    const float a0 = a_edge[0], a1 = a_edge[1], a2 = a_edge[2];
    __syncthreads();

    const int j = t & 15, i0 = t >> 4;                // 8 i-stripes x 16 j
    for (int i = i0; i < NN; i += 8) {
        float base = sEs[i] + sEd[j] + distance[i * NN + j0 + j] * a0;
        float x1 = base + a1; x1 = x1 > 0.f ? x1 : 0.2f * x1;
        float x2 = base + a2; x2 = x2 > 0.f ? x2 : 0.2f * x2;
        float s1 = adj_sta[i * NN + j0 + j] ? x1 : -30000.0f;
        float s2 = adj_req[((size_t)b * NN + i) * NN + j0 + j] ? x2 : -30000.0f;
        sS[i * 16 + j] = (_Float16)s1;
        sR[i * 16 + j] = (_Float16)s2;
    }
    __syncthreads();

    float m = -1e30f;                                  // joint max over 2N
    for (int i = i0; i < NN; i += 8)
        m = fmaxf(m, fmaxf((float)sS[i * 16 + j], (float)sR[i * 16 + j]));
    red[j * 8 + i0] = m;
    __syncthreads();
    if (i0 == 0) {
        float mm = red[j * 8];
        for (int u = 1; u < 8; ++u) mm = fmaxf(mm, red[j * 8 + u]);
        red[j * 8] = mm;
    }
    __syncthreads();
    m = red[j * 8];
    __syncthreads();

    float ssum = 0.f, wloc[32];
    int cnt = 0;
    for (int i = i0; i < NN; i += 8) {
        float w = __expf((float)sS[i * 16 + j] - m) + __expf((float)sR[i * 16 + j] - m);
        wloc[cnt++] = w; ssum += w;
    }
    red[j * 8 + i0] = ssum;
    __syncthreads();
    if (i0 == 0) {
        float s2 = 0.f;
        for (int u = 0; u < 8; ++u) s2 += red[j * 8 + u];
        red[j * 8] = s2;
    }
    __syncthreads();
    const float inv = 1.0f / red[j * 8];
    cnt = 0;
    for (int i = i0; i < NN; i += 8)
        sWT[j * NNP + i] = (_Float16)(wloc[cnt++] * inv);
    __syncthreads();

    // Aggregation GEMM: 16(j) x 64(cols) x 256(K). 4 waves, one 16-col slab each.
    const int lane = t & 31, wv = t >> 5;
    const int hi = lane >> 4, lm = lane & 15;
    v8f c = {};
    for (int k0 = 0; k0 < NN; k0 += 32) {
        const v16h a  = load_a_frag(sWT, NNP, lm, hi, k0);
        const v16h bf = load_b_frag(sK + k0 * 64 + wv * 16, 64, lm, hi);
        c = __builtin_amdgcn_wmma_f32_16x16x32_f16(false, a, false, bf,
                                                   (short)0, c, false, false);
    }
#pragma unroll
    for (int v = 0; v < 8; ++v) {
        float x = c[v];
        x = x > 0.f ? x : (__expf(x) - 1.0f);          // ELU
        const size_t r = rowb + j0 + v + (hi << 3);
        const int cc = wv * 16 + lm;
        out1[r * ldo1 + off1 + cc] = (_Float16)x;
        if (out2) out2[r * ldo2 + off2 + cc] = (_Float16)x;
    }
}

// ------------------------------- small kernels -----------------------------
__global__ void pack_w(const float* __restrict__ src, int srcLd, int srcOff,
                       _Float16* __restrict__ dst, int dstLd, int rows, int cols)
{
    int i = blockIdx.x * 256 + threadIdx.x;
    if (i >= rows * cols) return;
    int r = i / cols, c = i - r * cols;
    dst[(size_t)r * dstLd + c] = (_Float16)src[(size_t)r * srcLd + srcOff + c];
}

__global__ void f32_to_f16(const float* __restrict__ src, _Float16* __restrict__ dst, int n)
{
    int i = blockIdx.x * 256 + threadIdx.x;
    if (i < n) dst[i] = (_Float16)src[i];
}

__global__ void make_gru_bias(const float* bih, const float* bhh, float* dst)
{
    int i = blockIdx.x * 128 + threadIdx.x;
    if (i < 384) dst[i] = bih[i] + (i < 256 ? bhh[i] : 0.0f);
}

__global__ __launch_bounds__(128)
void build_observe(const float* __restrict__ obs_feats, const int* __restrict__ time_idx,
                   const int* __restrict__ tp_idx, const int* __restrict__ cs_idx,
                   const float* __restrict__ action_pre,
                   const float* __restrict__ time_emb, const float* __restrict__ cs_emb,
                   const float* __restrict__ tp_emb,
                   _Float16* obsB, _Float16* obsF, _Float16* xm)
{
    int row = blockIdx.x * 128 + threadIdx.x;
    if (row >= BNROWS) return;
    int n = row & (NN - 1);
    float ob[15];
    int cs = cs_idx[n];
#pragma unroll
    for (int k = 0; k < 4; ++k) ob[k] = cs_emb[cs * 4 + k];
    int tp = tp_idx[row];
    ob[4] = tp_emb[tp * 2]; ob[5] = tp_emb[tp * 2 + 1];
#pragma unroll
    for (int k = 0; k < 9; ++k) ob[6 + k] = obs_feats[(size_t)row * 9 + k];
    _Float16* pb = obsB + (size_t)row * 32;
#pragma unroll
    for (int k = 0; k < 15; ++k) pb[k] = (_Float16)ob[k];
    _Float16* pf = obsF + (size_t)row * 160;
    int tt = time_idx[row];
#pragma unroll
    for (int k = 0; k < 4; ++k) pf[k] = (_Float16)time_emb[tt * 4 + k];
    _Float16* px = xm + (size_t)row * 96;
#pragma unroll
    for (int k = 0; k < 14; ++k) { pf[4 + k] = (_Float16)ob[k]; px[k] = (_Float16)ob[k]; }
    pf[146] = (_Float16)action_pre[row];
}

__global__ __launch_bounds__(32)
void post_qk(const float* __restrict__ qk, const float* __restrict__ a_src,
             const float* __restrict__ a_dst, float* e_src, float* e_dst,
             _Float16* __restrict__ kh)
{
    const int row = blockIdx.x, l = threadIdx.x;
    const float* q = qk + (size_t)row * 128;
    float k0 = q[l], k1 = q[l + 32];
    float s = k0 * a_src[l] + k1 * a_src[l + 32];
    float d = q[64 + l] * a_dst[l] + q[96 + l] * a_dst[l + 32];
    kh[(size_t)row * 64 + l]      = (_Float16)k0;
    kh[(size_t)row * 64 + l + 32] = (_Float16)k1;
    for (int off = 16; off; off >>= 1) {
        s += __shfl_down(s, off, 32);
        d += __shfl_down(d, off, 32);
    }
    if (l == 0) { e_src[row] = s; e_dst[row] = d; }
}

__global__ __launch_bounds__(256)
void gru_gate(const float* __restrict__ gi, const float* __restrict__ ghn,
              const float* __restrict__ h_pre, float* __restrict__ h_out,
              _Float16* __restrict__ hth)
{
    int idx = blockIdx.x * 256 + threadIdx.x;          // BN*128 total
    int row = idx >> 7, h = idx & 127;
    const float* g = gi + (size_t)row * 384;
    float r = 1.0f / (1.0f + __expf(-g[h]));
    float z = 1.0f / (1.0f + __expf(-g[128 + h]));
    float nn = tanhf(g[256 + h] + r * ghn[(size_t)row * 128 + h]);
    float ht = (1.0f - z) * nn + z * h_pre[idx];
    h_out[idx] = ht;
    hth[idx] = (_Float16)ht;
}

__global__ __launch_bounds__(32)
void final_act(const _Float16* __restrict__ common, const _Float16* __restrict__ w,
               const _Float16* __restrict__ bt2, const float* __restrict__ bw3,
               const float* __restrict__ bb3, float* __restrict__ out)
{
    const int row = blockIdx.x, l = threadIdx.x;
    const _Float16* c = common + (size_t)row * 128;
    const _Float16* ww = w + (size_t)row * 128;
    float dot = 0.0f;
#pragma unroll
    for (int u = 0; u < 4; ++u) dot += (float)c[l + 32 * u] * (float)ww[l + 32 * u];
    float v = dot + ((l < 16) ? (float)bt2[(size_t)row * 32 + l] * bw3[l] : 0.0f);
    for (int off = 16; off; off >>= 1) v += __shfl_down(v, off, 32);
    if (l == 0) out[row] = 1.0f / (1.0f + __expf(-(v + bb3[0])));
}

// ---------------------------------------------------------------------------
extern "C" void kernel_launch(void* const* d_in, const int* in_sizes, int n_in,
                              void* d_out, int out_size, void* d_ws, size_t ws_size,
                              hipStream_t stream)
{
    (void)in_sizes; (void)n_in; (void)out_size; (void)ws_size;
    const float* obs_feats  = (const float*)d_in[0];
    const int*   time_idx   = (const int*)d_in[1];
    const int*   tp_idx     = (const int*)d_in[2];
    const int*   cs_idx     = (const int*)d_in[3];
    const int*   adj_c_dist = (const int*)d_in[4];
    const int*   adj_c_req  = (const int*)d_in[5];
    const int*   adj_o_dist = (const int*)d_in[6];
    const int*   adj_o_req  = (const int*)d_in[7];
    const float* distance   = (const float*)d_in[8];
    const float* h_pre      = (const float*)d_in[9];
    const float* action_pre = (const float*)d_in[10];
    const float* time_emb   = (const float*)d_in[11];
    const float* cs_emb     = (const float*)d_in[12];
    const float* tp_emb     = (const float*)d_in[13];
    const float* comp_Wa    = (const float*)d_in[14];
    const float* comp_Wb    = (const float*)d_in[15];
    const float* comp_asrc  = (const float*)d_in[16];
    const float* comp_adst  = (const float*)d_in[17];
    const float* comp_aedge = (const float*)d_in[18];
    const float* coop_Wa    = (const float*)d_in[19];
    const float* coop_Wb    = (const float*)d_in[20];
    const float* coop_asrc  = (const float*)d_in[21];
    const float* coop_adst  = (const float*)d_in[22];
    const float* coop_aedge = (const float*)d_in[23];
    const float* actor_W1   = (const float*)d_in[24];
    const float* actor_b1   = (const float*)d_in[25];
    const float* actor_W2   = (const float*)d_in[26];
    const float* actor_b2   = (const float*)d_in[27];
    const float* gru_Wih    = (const float*)d_in[28];
    const float* gru_Whh    = (const float*)d_in[29];
    const float* gru_bih    = (const float*)d_in[30];
    const float* gru_bhh    = (const float*)d_in[31];
    const float* mw1 = (const float*)d_in[32]; const float* mb1 = (const float*)d_in[33];
    const float* mw2 = (const float*)d_in[34]; const float* mb2 = (const float*)d_in[35];
    const float* mw3 = (const float*)d_in[36]; const float* mb3 = (const float*)d_in[37];
    const float* bw1 = (const float*)d_in[38]; const float* bb1 = (const float*)d_in[39];
    const float* bw2 = (const float*)d_in[40]; const float* bb2 = (const float*)d_in[41];
    const float* bw3 = (const float*)d_in[42]; const float* bb3 = (const float*)d_in[43];

    char* ws = (char*)d_ws;
    _Float16* obsB  = (_Float16*)(ws + OFF_OBS_B);
    _Float16* xm    = (_Float16*)(ws + OFF_XM);
    _Float16* obsF  = (_Float16*)(ws + OFF_OBSF);
    _Float16* mt2   = (_Float16*)(ws + OFF_MT2);
    _Float16* bt2   = (_Float16*)(ws + OFF_BT2);
    _Float16* Wcomp = (_Float16*)(ws + OFF_WCOMP);
    _Float16* Wcoop = (_Float16*)(ws + OFF_WCOOP);
    _Float16* W1h   = (_Float16*)(ws + OFF_W1H);
    _Float16* W2h   = (_Float16*)(ws + OFF_W2H);
    _Float16* Wih   = (_Float16*)(ws + OFF_WIH);
    _Float16* Whrz  = (_Float16*)(ws + OFF_WHRZ);
    _Float16* Whn   = (_Float16*)(ws + OFF_WHN);
    _Float16* Mw1   = (_Float16*)(ws + OFF_MW1);
    _Float16* Mw2   = (_Float16*)(ws + OFF_MW2);
    _Float16* Mw3   = (_Float16*)(ws + OFF_MW3);
    _Float16* Bw1   = (_Float16*)(ws + OFF_BW1);
    _Float16* Bw2   = (_Float16*)(ws + OFF_BW2);
    float*    biasc = (float*)(ws + OFF_BIASC);
    float*    gi    = (float*)(ws + OFF_GI);     // also qk scratch [BN x 128]
    float*    ghn   = (float*)(ws + OFF_GHN);
    float*    es1   = (float*)(ws + OFF_ES1);
    float*    ed1   = (float*)(ws + OFF_ED1);
    float*    es2   = (float*)(ws + OFF_ES2);
    float*    ed2   = (float*)(ws + OFF_ED2);
    _Float16* k1h   = (_Float16*)(ws + OFF_K1H);
    _Float16* k2h   = (_Float16*)(ws + OFF_K2H);
    _Float16* hpreh = (_Float16*)(ws + OFF_HPREH);
    _Float16* h1h   = (_Float16*)(ws + OFF_H1H);
    _Float16* comh  = (_Float16*)(ws + OFF_COMH);
    _Float16* hth   = (_Float16*)(ws + OFF_HTH);
    _Float16* mt1   = (_Float16*)(ws + OFF_MT1);
    _Float16* bt1   = (_Float16*)(ws + OFF_BT1);
    _Float16* wh    = (_Float16*)(ws + OFF_WH);

    float* out_act = (float*)d_out;              // [BN]
    float* out_ht  = (float*)d_out + BNROWS;     // [BN x 128]

    // zero padded-activation + weight arenas (graph-capture safe memset node)
    hipMemsetAsync(ws, 0, OFF_W_END, stream);

    auto PACK = [&](const float* s, int sld, int soff, _Float16* d, int dld, int r, int c) {
        pack_w<<<(r * c + 255) / 256, 256, 0, stream>>>(s, sld, soff, d, dld, r, c);
    };
    PACK(comp_Wb, 64, 0, Wcomp,      128, 15, 64);
    PACK(comp_Wa, 64, 0, Wcomp + 64, 128, 14, 64);
    PACK(coop_Wb, 64, 0, Wcoop,      128, 78, 64);
    PACK(coop_Wa, 64, 0, Wcoop + 64, 128, 78, 64);
    PACK(actor_W1, 128, 0, W1h, 128, 146, 128);
    PACK(actor_W2, 128, 0, W2h, 128, 128, 128);
    PACK(gru_Wih, 384, 0, Wih, 384, 147, 384);
    PACK(gru_Whh, 384, 0,   Whrz, 256, 128, 256);
    PACK(gru_Whh, 384, 256, Whn,  128, 128, 128);
    PACK(mw1, 32, 0, Mw1, 32, 128, 32);
    PACK(mw2, 16, 0, Mw2, 16, 32, 16);
    PACK(mw3, 128, 0, Mw3, 128, 16, 128);
    PACK(bw1, 32, 0, Bw1, 32, 128, 32);
    PACK(bw2, 16, 0, Bw2, 16, 32, 16);
    make_gru_bias<<<3, 128, 0, stream>>>(gru_bih, gru_bhh, biasc);
    f32_to_f16<<<(BNROWS * 128) / 256, 256, 0, stream>>>(h_pre, hpreh, BNROWS * 128);

    build_observe<<<BNROWS / 128, 128, 0, stream>>>(
        obs_feats, time_idx, tp_idx, cs_idx, action_pre,
        time_emb, cs_emb, tp_emb, obsB, obsF, xm);

    auto GEMM = [&](const _Float16* A, int lda, const _Float16* W, int ldw,
                    const float* bias, int K, int nout, float* oF, int ldof, int acc,
                    _Float16* oH, int ldoh, int offH, int act) {
        gemm16<<<dim3(BNROWS / 16, nout / 16), 32, 0, stream>>>(
            A, lda, W, ldw, bias, K, oF, ldof, acc, oH, ldoh, offH, act);
    };

    // --- comp GAT ---
    GEMM(obsB, 32, Wcomp, 128, nullptr, 32, 128, gi, 128, 0, nullptr, 0, 0, 0);
    post_qk<<<BNROWS, 32, 0, stream>>>(gi, comp_asrc, comp_adst, es1, ed1, k1h);
    gat_agg<<<dim3(NN / 16, NN), 128, 0, stream>>>(
        es1, ed1, distance, comp_aedge, adj_c_dist, adj_c_req, k1h,
        obsF, 160, 18, xm, 96, 14);

    // --- coop GAT ---
    GEMM(xm, 96, Wcoop, 128, nullptr, 96, 128, gi, 128, 0, nullptr, 0, 0, 0);
    post_qk<<<BNROWS, 32, 0, stream>>>(gi, coop_asrc, coop_adst, es2, ed2, k2h);
    gat_agg<<<dim3(NN / 16, NN), 128, 0, stream>>>(
        es2, ed2, distance, coop_aedge, adj_o_dist, adj_o_req, k2h,
        obsF, 160, 82, nullptr, 0, 0);

    // --- actor MLP ---
    GEMM(obsF, 160, W1h, 128, actor_b1, 160, 128, nullptr, 0, 0, h1h, 128, 0, 1);
    GEMM(h1h, 128, W2h, 128, actor_b2, 128, 128, nullptr, 0, 0, comh, 128, 0, 1);

    // --- GRU ---
    GEMM(obsF, 160, Wih, 384, biasc, 160, 384, gi, 384, 0, nullptr, 0, 0, 0);
    GEMM(hpreh, 128, Whrz, 256, nullptr, 128, 256, gi, 384, 1, nullptr, 0, 0, 0);
    GEMM(hpreh, 128, Whn, 128, gru_bhh + 256, 128, 128, ghn, 128, 0, nullptr, 0, 0, 0);
    gru_gate<<<(BNROWS * 128) / 256, 256, 0, stream>>>(gi, ghn, h_pre, out_ht, hth);

    // --- meta hypernetwork ---
    GEMM(hth, 128, Mw1, 32, mb1, 128, 32, nullptr, 0, 0, mt1, 32, 0, 2);
    GEMM(mt1, 32, Mw2, 16, mb2, 32, 16, nullptr, 0, 0, mt2, 32, 0, 2);
    GEMM(mt2, 32, Mw3, 128, mb3, 32, 128, nullptr, 0, 0, wh, 128, 0, 0);
    GEMM(hth, 128, Bw1, 32, bb1, 128, 32, nullptr, 0, 0, bt1, 32, 0, 2);
    GEMM(bt1, 32, Bw2, 16, bb2, 32, 16, nullptr, 0, 0, bt2, 32, 0, 2);

    final_act<<<BNROWS, 32, 0, stream>>>(comh, wh, bt2, bw3, bb3, out_act);
}
